// TinyVitBlockSW_25185688223960
// MI455X (gfx1250) — matrix operand order repaired and checked by reading, not compiled, not run on hardware
//
#include <hip/hip_runtime.h>
#include <hip/hip_bf16.h>
#include <math.h>

typedef _Float16 half_t;
typedef __attribute__((ext_vector_type(16))) _Float16 v16h;
typedef __attribute__((ext_vector_type(8)))  _Float16 v8h;
typedef __attribute__((ext_vector_type(8)))  float    v8f;

#define BATCH   64
#define HDIM    28
#define WDIM    28
#define CDIM    384
#define NHEADS  12
#define HD      32
#define WSZ     7
#define NTOK    49                // tokens per window (7*7)
#define NTOKP   64                // padded tokens
#define NWS     4                 // windows per side (28/7)
#define NWIN    (BATCH*NWS*NWS)   // 1024
#define HIDDEN  (4*CDIM)          // 1536
#define NPIX    (HDIM*WDIM)       // 784
#define NTOKENS (BATCH*NPIX)      // 50176
#define EPSF    1e-5f
#define QSCALE  0.1767766953f     // 32^-0.5
#define PLANE   (NTOKP*HD)        // 2048 halves per q/k/v plane

// ---------------- WMMA fragment helpers (wave32, 16x16x32 f16->f32) --------

__device__ __forceinline__ v8f wmma16(v16h a, v16h b, v8f c) {
  return __builtin_amdgcn_wmma_f32_16x16x32_f16(false, a, false, b,
                                                (short)0, c, false, false);
}

__device__ __forceinline__ v16h combine16(v8h lo, v8h hi) {
  return __builtin_shufflevector(lo, hi, 0, 1, 2, 3, 4, 5, 6, 7,
                                 8, 9, 10, 11, 12, 13, 14, 15);
}

// A: 16x32 (MxK) f16 fragment, row-major, NO bounds checks (padded input).
// lane = hf*16 + m; per lane two contiguous 16B runs at k = hf*8 and 16+hf*8.
// Requires: row base and kBase offsets 16-byte aligned (lda even multiple of 8).
__device__ __forceinline__ v16h load_a_nc(const half_t* A, int lda,
                                          int mBase, int kBase, int lane) {
  const int hf  = lane >> 4;
  const int row = mBase + (lane & 15);
  const half_t* p = A + (size_t)row * lda + kBase + hf * 8;
  v8h lo = *(const v8h*)p;
  v8h hi = *(const v8h*)(p + 16);
  return combine16(lo, hi);
}

// B = W^T where W is [N][ldw] row-major (x @ W.T): element (k,n) = W[n][k].
// lane = hf*16 + n; per lane two contiguous 16B runs at k = kBase + hf*8 and kBase + 16 + hf*8.
__device__ __forceinline__ v16h load_bT_nc(const half_t* W, int ldw,
                                           int kBase, int nBase, int lane) {
  const int hf  = lane >> 4;
  const int col = nBase + (lane & 15);
  const half_t* p = W + (size_t)col * ldw + kBase + hf * 8;
  v8h lo = *(const v8h*)p;
  v8h hi = *(const v8h*)(p + 16);
  return combine16(lo, hi);
}

// C/D f32 16x16: lane = hf*16 + n ; VGPR r: m = r + 8*hf. Unconditional.
__device__ __forceinline__ void store_c_nc(float* C, int ldc,
                                           int mBase, int nBase,
                                           v8f acc, int lane) {
  const int hf  = lane >> 4;
  const int col = nBase + (lane & 15);
#pragma unroll
  for (int r = 0; r < 8; ++r)
    C[(size_t)(mBase + r + 8 * hf) * ldc + col] = acc[r];
}

// ---------------- K0: fp32 -> f16 weight conversion ------------------------

__global__ __launch_bounds__(256)
void k_f32_to_f16(const float* __restrict__ src, half_t* __restrict__ dst, int n) {
  int i = blockIdx.x * 256 + threadIdx.x;
  if (i < n) dst[i] = (half_t)src[i];
}

// ---------------- K0b: expand relative-position bias table -----------------

__global__ __launch_bounds__(256)
void k_bias_tab(const float* __restrict__ biases, const int* __restrict__ idxs,
                int n_off, float* __restrict__ tab) {
  int i = blockIdx.x * 256 + threadIdx.x;
  if (i < NHEADS * NTOK * NTOK) {
    const int h  = i / (NTOK * NTOK);
    const int ij = i % (NTOK * NTOK);
    tab[i] = biases[(size_t)h * n_off + idxs[ij]];
  }
}

// ---------------- K1: window partition + LN1 + QKV GEMM --------------------
// one block per window (256 threads = 8 waves)

__global__ __launch_bounds__(256)
void k_ln_qkv(const float* __restrict__ x,
              const float* __restrict__ g, const float* __restrict__ bvec,
              const half_t* __restrict__ qkv_w16,
              const float* __restrict__ qkv_b,
              half_t* __restrict__ qkv16) {
  __shared__ half_t hn[NTOKP][CDIM + 8];
  const int win = blockIdx.x;
  const int bb  = win / (NWS * NWS);
  const int wh  = (win / NWS) % NWS;
  const int ww  = win % NWS;

  // zero pad rows 49..63 (so A-fragment loads need no bounds checks)
  for (int i = threadIdx.x; i < (NTOKP - NTOK) * (CDIM + 8); i += 256) {
    const int t = NTOK + i / (CDIM + 8);
    const int c = i % (CDIM + 8);
    hn[t][c] = (half_t)0.0f;
  }
  // LayerNorm per token (fp32 -> f16 into LDS)
  for (int t = threadIdx.x; t < NTOK; t += 256) {
    const int hh = wh * WSZ + t / WSZ;
    const int wc = ww * WSZ + t % WSZ;
    const float* xr = x + (((size_t)bb * HDIM + hh) * WDIM + wc) * CDIM;
    float s = 0.f, s2 = 0.f;
    for (int c = 0; c < CDIM; ++c) { float v = xr[c]; s += v; s2 += v * v; }
    const float mean = s / CDIM;
    const float inv  = rsqrtf(s2 / CDIM - mean * mean + EPSF);
    for (int c = 0; c < CDIM; ++c)
      hn[t][c] = (half_t)((xr[c] - mean) * inv * g[c] + bvec[c]);
  }
  __syncthreads();

  const int wave = threadIdx.x >> 5, lane = threadIdx.x & 31;
  const int hf = lane >> 4;
  // 4 M-tiles x 72 N-tiles
  for (int tile = wave; tile < 4 * 72; tile += 8) {
    const int mBase = (tile % 4) * 16;
    const int nBase = (tile / 4) * 16;
    v8f acc = {};
#pragma unroll 4
    for (int ks = 0; ks < CDIM / 32; ++ks) {
      v16h a = load_a_nc(&hn[0][0], CDIM + 8, mBase, ks * 32, lane);
      v16h b = load_bT_nc(qkv_w16, CDIM, ks * 32, nBase, lane);
      acc = wmma16(a, b, acc);
    }
    const int col   = nBase + (lane & 15);
    const int head  = col / (3 * HD);
    const int j     = col % (3 * HD);
    const int which = j / HD;
    const int d     = j % HD;
    const float bias = qkv_b[col];
    const float scl  = (which == 0) ? QSCALE : 1.0f;
    half_t* plane = qkv16 + (((size_t)win * NHEADS + head) * 3 + which) * PLANE;
#pragma unroll
    for (int r = 0; r < 8; ++r) {
      const int row = mBase + r + 8 * hf;
      // pad rows must be exact zero (bias would pollute them otherwise)
      const float val = (row < NTOK) ? (acc[r] + bias) * scl : 0.0f;
      if (which == 2) plane[d * NTOKP + row] = (half_t)val;  // V stored transposed
      else            plane[row * HD + d]    = (half_t)val;
    }
  }
}

// ---------------- K2: windowed attention per (win, head) -------------------
// block = 128 threads (4 waves)

__global__ __launch_bounds__(128)
void k_attn(const half_t* __restrict__ qkv16,
            const float* __restrict__ biasTab,
            half_t* __restrict__ o16) {
  __shared__ float  S[NTOKP][NTOKP + 1];
  __shared__ half_t P[NTOKP][NTOKP];
  const int win  = blockIdx.x / NHEADS;
  const int head = blockIdx.x % NHEADS;
  const half_t* q  = qkv16 + (((size_t)win * NHEADS + head) * 3 + 0) * PLANE;
  const half_t* k  = qkv16 + (((size_t)win * NHEADS + head) * 3 + 1) * PLANE;
  const half_t* vT = qkv16 + (((size_t)win * NHEADS + head) * 3 + 2) * PLANE;
  const int wave = threadIdx.x >> 5, lane = threadIdx.x & 31;

  // zero P (covers padded rows/cols)
  for (int i = threadIdx.x; i < NTOKP * NTOKP; i += 128)
    (&P[0][0])[i] = (half_t)0.0f;

  // S = q(scaled) @ k^T : 4x4 tiles, K=32. Padded rows/cols of q,k are zero.
  for (int tile = wave; tile < 16; tile += 4) {
    const int mBase = (tile & 3) * 16, nBase = (tile >> 2) * 16;
    v16h a = load_a_nc(q, HD, mBase, 0, lane);
    v16h b = load_bT_nc(k, HD, 0, nBase, lane);
    v8f acc = {};
    acc = wmma16(a, b, acc);
    store_c_nc(&S[0][0], NTOKP + 1, mBase, nBase, acc, lane);
  }
  __syncthreads();

  // bias + softmax over valid 49x49 region, in place in LDS
  for (int i = threadIdx.x; i < NTOK; i += 128) {
    const float* bt = biasTab + ((size_t)head * NTOK + i) * NTOK;
    float mx = -1e30f;
    for (int j = 0; j < NTOK; ++j) {
      const float s = S[i][j] + bt[j];
      S[i][j] = s;
      mx = fmaxf(mx, s);
    }
    float sum = 0.f;
    for (int j = 0; j < NTOK; ++j) {
      const float e = __expf(S[i][j] - mx);
      S[i][j] = e;
      sum += e;
    }
    const float rs = 1.0f / sum;
    for (int j = 0; j < NTOK; ++j) P[i][j] = (half_t)(S[i][j] * rs);
  }
  __syncthreads();

  // O = P @ V : 4 M-tiles x 2 N-tiles, K=64. V stored transposed [32][64].
  const int hf = lane >> 4;
  for (int tile = wave; tile < 8; tile += 4) {
    const int mBase = (tile >> 1) * 16, nBase = (tile & 1) * 16;
    v8f acc = {};
#pragma unroll
    for (int ks = 0; ks < 2; ++ks) {
      v16h a = load_a_nc(&P[0][0], NTOKP, mBase, ks * 32, lane);
      v16h b = load_bT_nc(vT, NTOKP, ks * 32, nBase, lane);  // (k,n)=vT[n][k]
      acc = wmma16(a, b, acc);
    }
    const int col = nBase + (lane & 15);
#pragma unroll
    for (int r = 0; r < 8; ++r) {
      const int row = mBase + r + 8 * hf;   // pad rows of P are zero -> acc 0
      o16[((size_t)win * NTOKP + row) * CDIM + head * HD + col] = (half_t)acc[r];
    }
  }
}

// ---------------- K3: output projection + residual (un-window) -------------

__global__ __launch_bounds__(256)
void k_proj(const half_t* __restrict__ o16,
            const half_t* __restrict__ proj_w16,
            const float* __restrict__ proj_b,
            const float* __restrict__ x,
            float* __restrict__ X1) {
  __shared__ half_t ot[NTOKP][CDIM + 8];
  const int win = blockIdx.x;
  const int bb  = win / (NWS * NWS);
  const int wh  = (win / NWS) % NWS;
  const int ww  = win % NWS;

  for (int i = threadIdx.x; i < NTOKP * CDIM; i += 256)
    ot[i / CDIM][i % CDIM] = o16[(size_t)win * NTOKP * CDIM + i];
  __syncthreads();

  const int wave = threadIdx.x >> 5, lane = threadIdx.x & 31;
  const int hf = lane >> 4;
  for (int tile = wave; tile < 4 * 24; tile += 8) {
    const int mBase = (tile % 4) * 16;
    const int nBase = (tile / 4) * 16;
    v8f acc = {};
#pragma unroll 4
    for (int ks = 0; ks < CDIM / 32; ++ks) {
      v16h a = load_a_nc(&ot[0][0], CDIM + 8, mBase, ks * 32, lane);
      v16h b = load_bT_nc(proj_w16, CDIM, ks * 32, nBase, lane);
      acc = wmma16(a, b, acc);
    }
    const int col = nBase + (lane & 15);
    const float pb = proj_b[col];
#pragma unroll
    for (int r = 0; r < 8; ++r) {
      const int row = mBase + r + 8 * hf;
      if (row < NTOK) {
        const int hh = wh * WSZ + row / WSZ;
        const int wc = ww * WSZ + row % WSZ;
        const size_t gi = (((size_t)bb * HDIM + hh) * WDIM + wc) * CDIM + col;
        X1[gi] = x[gi] + acc[r] + pb;
      }
    }
  }
}

// ---------------- K4: depthwise 3x3 conv + BN ------------------------------

__global__ __launch_bounds__(256)
void k_dwconv_bn(const float* __restrict__ X1,
                 const float* __restrict__ conv_w,
                 const float* __restrict__ bn_g, const float* __restrict__ bn_b,
                 const float* __restrict__ bn_mean, const float* __restrict__ bn_var,
                 float* __restrict__ XT) {
  const size_t idx = (size_t)blockIdx.x * 256 + threadIdx.x;
  if (idx >= (size_t)NTOKENS * CDIM) return;
  const int c = (int)(idx % CDIM);
  const int p = (int)(idx / CDIM);
  const int w = p % WDIM;
  const int h = (p / WDIM) % HDIM;
  const int b = p / NPIX;
  float acc = 0.f;
#pragma unroll
  for (int ky = 0; ky < 3; ++ky) {
    const int hh = h + ky - 1;
    if (hh < 0 || hh >= HDIM) continue;
#pragma unroll
    for (int kx = 0; kx < 3; ++kx) {
      const int wc = w + kx - 1;
      if (wc < 0 || wc >= WDIM) continue;
      acc += X1[(((size_t)b * HDIM + hh) * WDIM + wc) * CDIM + c] *
             conv_w[c * 9 + ky * 3 + kx];
    }
  }
  const float inv = rsqrtf(bn_var[c] + EPSF);
  XT[idx] = (acc - bn_mean[c]) * (bn_g[c] * inv) + bn_b[c];
}

// ---------------- K5: LN2 -> f16 tokens ------------------------------------

__global__ __launch_bounds__(256)
void k_ln2(const float* __restrict__ XT,
           const float* __restrict__ g, const float* __restrict__ bvec,
           half_t* __restrict__ M16) {
  const int t = blockIdx.x * 256 + threadIdx.x;
  if (t >= NTOKENS) return;
  const float* xr = XT + (size_t)t * CDIM;
  float s = 0.f, s2 = 0.f;
  for (int c = 0; c < CDIM; ++c) { float v = xr[c]; s += v; s2 += v * v; }
  const float mean = s / CDIM;
  const float inv  = rsqrtf(s2 / CDIM - mean * mean + EPSF);
  half_t* dst = M16 + (size_t)t * CDIM;
  for (int c = 0; c < CDIM; ++c)
    dst[c] = (half_t)((xr[c] - mean) * inv * g[c] + bvec[c]);
}

// ---------------- K6: FC1 + exact GELU -------------------------------------
// grid = (NTOKENS/64, HIDDEN/128), block = 256 (8 waves)

__global__ __launch_bounds__(256)
void k_fc1_gelu(const half_t* __restrict__ M16,
                const half_t* __restrict__ fc1_w16,
                const float* __restrict__ fc1_b,
                half_t* __restrict__ H1) {
  __shared__ half_t Ach[64][40];   // row stride 80B = 5*16B -> aligned b128
  const int tBase = blockIdx.x * 64;
  const int cBase = blockIdx.y * 128;
  const int wave = threadIdx.x >> 5, lane = threadIdx.x & 31;
  const int mt = wave & 3;
  const int ntBase = (wave >> 2) * 4;
  const int hf = lane >> 4;
  v8f acc[4] = {};

  for (int ks = 0; ks < CDIM / 32; ++ks) {
    __syncthreads();
    for (int i = threadIdx.x; i < 64 * 32; i += 256) {
      const int r = i >> 5, c = i & 31;
      Ach[r][c] = M16[(size_t)(tBase + r) * CDIM + ks * 32 + c];
    }
    __syncthreads();
    v16h a = load_a_nc(&Ach[0][0], 40, mt * 16, 0, lane);
#pragma unroll
    for (int i = 0; i < 4; ++i) {
      const int nBase = cBase + (ntBase + i) * 16;
      v16h b = load_bT_nc(fc1_w16, CDIM, ks * 32, nBase, lane);
      acc[i] = wmma16(a, b, acc[i]);
    }
  }
#pragma unroll
  for (int i = 0; i < 4; ++i) {
    const int nBase = cBase + (ntBase + i) * 16;
    const int col = nBase + (lane & 15);
    const float fb = fc1_b[col];
#pragma unroll
    for (int r = 0; r < 8; ++r) {
      const int row = tBase + mt * 16 + r + 8 * hf;
      const float val = acc[i][r] + fb;
      const float ge = 0.5f * val * (1.0f + erff(val * 0.70710678118654752f));
      H1[(size_t)row * HIDDEN + col] = (half_t)ge;
    }
  }
}

// ---------------- K7: FC2 + residual -> out --------------------------------
// grid = (NTOKENS/64, CDIM/128), block = 256 (8 waves)

__global__ __launch_bounds__(256)
void k_fc2_res(const half_t* __restrict__ H1,
               const half_t* __restrict__ fc2_w16,
               const float* __restrict__ fc2_b,
               const float* __restrict__ XT,
               float* __restrict__ out) {
  __shared__ half_t Ach[64][40];
  const int tBase = blockIdx.x * 64;
  const int cBase = blockIdx.y * 128;
  const int wave = threadIdx.x >> 5, lane = threadIdx.x & 31;
  const int mt = wave & 3;
  const int ntBase = (wave >> 2) * 4;
  const int hf = lane >> 4;
  v8f acc[4] = {};

  for (int ks = 0; ks < HIDDEN / 32; ++ks) {
    __syncthreads();
    for (int i = threadIdx.x; i < 64 * 32; i += 256) {
      const int r = i >> 5, c = i & 31;
      Ach[r][c] = H1[(size_t)(tBase + r) * HIDDEN + ks * 32 + c];
    }
    __syncthreads();
    v16h a = load_a_nc(&Ach[0][0], 40, mt * 16, 0, lane);
#pragma unroll
    for (int i = 0; i < 4; ++i) {
      const int nBase = cBase + (ntBase + i) * 16;
      v16h b = load_bT_nc(fc2_w16, HIDDEN, ks * 32, nBase, lane);
      acc[i] = wmma16(a, b, acc[i]);
    }
  }
#pragma unroll
  for (int i = 0; i < 4; ++i) {
    const int nBase = cBase + (ntBase + i) * 16;
    const int col = nBase + (lane & 15);
    const float fb = fc2_b[col];
#pragma unroll
    for (int r = 0; r < 8; ++r) {
      const int row = tBase + mt * 16 + r + 8 * hf;
      const size_t gi = (size_t)row * CDIM + col;
      out[gi] = XT[gi] + acc[i][r] + fb;
    }
  }
}

// ---------------- launch ----------------------------------------------------

extern "C" void kernel_launch(void* const* d_in, const int* in_sizes, int n_in,
                              void* d_out, int out_size, void* d_ws, size_t ws_size,
                              hipStream_t stream) {
  const float* x        = (const float*)d_in[0];
  const float* ln1_g    = (const float*)d_in[1];
  const float* ln1_b    = (const float*)d_in[2];
  const float* qkv_w    = (const float*)d_in[3];
  const float* qkv_b    = (const float*)d_in[4];
  const float* proj_w   = (const float*)d_in[5];
  const float* proj_b   = (const float*)d_in[6];
  const float* attn_bia = (const float*)d_in[7];
  const float* conv_w   = (const float*)d_in[8];
  const float* bn_g     = (const float*)d_in[9];
  const float* bn_b     = (const float*)d_in[10];
  const float* bn_mean  = (const float*)d_in[11];
  const float* bn_var   = (const float*)d_in[12];
  const float* ln2_g    = (const float*)d_in[13];
  const float* ln2_b    = (const float*)d_in[14];
  const float* fc1_w    = (const float*)d_in[15];
  const float* fc1_b    = (const float*)d_in[16];
  const float* fc2_w    = (const float*)d_in[17];
  const float* fc2_b    = (const float*)d_in[18];
  const int*   bias_idx = (const int*)d_in[19];
  float* out = (float*)d_out;
  const int n_off = in_sizes[7] / NHEADS;

  char* ws = (char*)d_ws;
  size_t off = 0;
  auto alloc = [&](size_t bytes) {
    void* p = ws + off;
    off += (bytes + 255) & ~(size_t)255;
    return p;
  };
  half_t* qkv_w16  = (half_t*)alloc((size_t)3 * CDIM * CDIM * 2);
  half_t* proj_w16 = (half_t*)alloc((size_t)CDIM * CDIM * 2);
  half_t* fc1_w16  = (half_t*)alloc((size_t)HIDDEN * CDIM * 2);
  half_t* fc2_w16  = (half_t*)alloc((size_t)CDIM * HIDDEN * 2);
  float*  biasTab  = (float*)alloc((size_t)NHEADS * NTOK * NTOK * 4);
  const size_t actBase = off;
  half_t* qkv16 = (half_t*)alloc((size_t)NWIN * NHEADS * 3 * PLANE * 2);  // 151 MB
  half_t* o16   = (half_t*)alloc((size_t)NWIN * NTOKP * CDIM * 2);        // 50 MB
  float*  X1    = (float*)alloc((size_t)NTOKENS * CDIM * 4);
  float*  XT    = (float*)alloc((size_t)NTOKENS * CDIM * 4);
  half_t* M16   = (half_t*)alloc((size_t)NTOKENS * CDIM * 2);
  // H1 (NTOKENS*HIDDEN*2 = 154 MB) aliases the dead qkv16+o16 region (201 MB)
  half_t* H1 = (half_t*)(ws + actBase);

  // K0: weight conversions + bias table
  {
    int n;
    n = 3 * CDIM * CDIM;
    k_f32_to_f16<<<(n + 255) / 256, 256, 0, stream>>>(qkv_w, qkv_w16, n);
    n = CDIM * CDIM;
    k_f32_to_f16<<<(n + 255) / 256, 256, 0, stream>>>(proj_w, proj_w16, n);
    n = HIDDEN * CDIM;
    k_f32_to_f16<<<(n + 255) / 256, 256, 0, stream>>>(fc1_w, fc1_w16, n);
    n = CDIM * HIDDEN;
    k_f32_to_f16<<<(n + 255) / 256, 256, 0, stream>>>(fc2_w, fc2_w16, n);
    n = NHEADS * NTOK * NTOK;
    k_bias_tab<<<(n + 255) / 256, 256, 0, stream>>>(attn_bia, bias_idx, n_off, biasTab);
  }

  k_ln_qkv<<<NWIN, 256, 0, stream>>>(x, ln1_g, ln1_b, qkv_w16, qkv_b, qkv16);
  k_attn<<<NWIN * NHEADS, 128, 0, stream>>>(qkv16, biasTab, o16);
  k_proj<<<NWIN, 256, 0, stream>>>(o16, proj_w16, proj_b, x, X1);

  {
    const size_t n = (size_t)NTOKENS * CDIM;
    k_dwconv_bn<<<(unsigned)((n + 255) / 256), 256, 0, stream>>>(
        X1, conv_w, bn_g, bn_b, bn_mean, bn_var, XT);
  }
  k_ln2<<<(NTOKENS + 255) / 256, 256, 0, stream>>>(XT, ln2_g, ln2_b, M16);

  k_fc1_gelu<<<dim3(NTOKENS / 64, HIDDEN / 128), 256, 0, stream>>>(
      M16, fc1_w16, fc1_b, H1);
  k_fc2_res<<<dim3(NTOKENS / 64, CDIM / 128), 256, 0, stream>>>(
      H1, fc2_w16, fc2_b, XT, out);
}
